// MoELayer_17188459118823
// MI455X (gfx1250) — matrix operand order repaired and checked by reading, not compiled, not run on hardware
//
#include <hip/hip_runtime.h>
#include <hip/hip_bf16.h>

// Problem constants (from reference)
#define NB   2
#define NS   1024
#define HD   1024
#define NE   16
#define FD   2048
#define TOK  (NB*NS)            // 2048 tokens
#define PCAP (TOK + NE*32)      // 2560 padded-row capacity (32-row expert tiles)
#define MAXTILES (PCAP/32)      // 80 32-row tiles max
#define LBWF 0.01f

typedef __attribute__((ext_vector_type(16))) __bf16 v16bf;
typedef __attribute__((ext_vector_type(8)))  float  v8f;
typedef __attribute__((ext_vector_type(4)))  float  float4v;
typedef __attribute__((ext_vector_type(4)))  unsigned int uint4v;
typedef __attribute__((ext_vector_type(2)))  unsigned int uint2v;
typedef __attribute__((address_space(3))) unsigned short lds_ushort_t;

union BFU   { unsigned short u; __bf16 b; };
union Frag  { v16bf v; uint4v q[2]; };
union CFrag { v8f v; float e[8]; };
union PK2   { unsigned short s[2]; unsigned int u; };

__device__ inline unsigned short f2bf(float f) {
  BFU t; t.b = (__bf16)f;                 // native f32->bf16 convert (v_cvt_*bf16_f32)
  return t.u;
}
__device__ inline float geluf(float x) {
  return 0.5f * x * (1.0f + erff(x * 0.70710678118654752f));  // exact erf gelu
}

// ---------------- K0: init ----------------
__global__ void moe_init(int* counts, int* cursor, int* perm) {
  int t = threadIdx.x + blockIdx.x * blockDim.x;
  if (t < NE) { counts[t] = 0; cursor[t] = 0; }
  for (int p = t; p < PCAP; p += blockDim.x * gridDim.x) perm[p] = -1;
}

// ---------------- K1: router (one wave32 per token) ----------------
__global__ void moe_router(const float* __restrict__ x, const float* __restrict__ rw,
                           float* __restrict__ topw, int* __restrict__ topidx,
                           int* __restrict__ counts, float* __restrict__ probs_out) {
  int wave = threadIdx.x >> 5;
  int lane = threadIdx.x & 31;
  int t = blockIdx.x * 4 + wave;
  float acc[NE];
#pragma unroll
  for (int e = 0; e < NE; e++) acc[e] = 0.f;
  const float* xr = x + (size_t)t * HD;
  for (int i = lane; i < HD; i += 32) {
    float xv = xr[i];
    const float* r = rw + (size_t)i * NE;
#pragma unroll
    for (int e = 0; e < NE; e++) acc[e] += xv * r[e];
  }
#pragma unroll
  for (int e = 0; e < NE; e++)
    for (int off = 16; off >= 1; off >>= 1) acc[e] += __shfl_xor(acc[e], off, 32);
  if (lane == 0) {
    float mx = acc[0];
#pragma unroll
    for (int e = 1; e < NE; e++) mx = fmaxf(mx, acc[e]);
    float p[NE]; float s = 0.f;
#pragma unroll
    for (int e = 0; e < NE; e++) { p[e] = expf(acc[e] - mx); s += p[e]; }
    float inv = 1.f / s;
    int bi = 0; float bp = p[0] * inv; p[0] = bp;
#pragma unroll
    for (int e = 1; e < NE; e++) {
      p[e] *= inv;
      if (p[e] > bp) { bp = p[e]; bi = e; }   // first max wins (argmax semantics)
    }
    topw[t] = bp; topidx[t] = bi;
    atomicAdd(&counts[bi], 1);
#pragma unroll
    for (int e = 0; e < NE; e++) probs_out[(size_t)t * NE + e] = p[e];
  }
}

// ---------------- K2: scan + 32-row tile metadata + scatter ----------------
__global__ void moe_scan(const int* __restrict__ counts, int* __restrict__ offsets,
                         int* __restrict__ tile_e, int* __restrict__ tile_r,
                         int* __restrict__ ntiles, const int* __restrict__ topidx,
                         int* __restrict__ cursor, int* __restrict__ perm) {
  if (threadIdx.x == 0) {
    int off = 0, nt = 0;
    for (int e = 0; e < NE; e++) {
      offsets[e] = off;
      int c = counts[e];
      int pt = (c + 31) >> 5;
      for (int i = 0; i < pt; i++) { tile_e[nt] = e; tile_r[nt] = off + i * 32; nt++; }
      off += pt * 32;
    }
    offsets[NE] = off;
    ntiles[0] = nt;
    for (int i = nt; i < MAXTILES; i++) { tile_e[i] = 0; tile_r[i] = 0; }
  }
  __syncthreads();
  for (int t = threadIdx.x; t < TOK; t += blockDim.x) {
    int e = topidx[t];
    int pos = offsets[e] + atomicAdd(&cursor[e], 1);
    perm[pos] = t;
  }
}

// ---------------- K3: gather tokens -> bf16 expert-sorted X (vectorized) ----------------
__global__ void moe_gather(const float* __restrict__ x, const int* __restrict__ perm,
                           unsigned short* __restrict__ Xg) {
  int p = blockIdx.x;
  int t = perm[p];
  int i = threadIdx.x * 4;                       // 256 threads * 4 = HD
  union { unsigned short s[4]; uint2v v; } u;
  if (t >= 0) {
    float4v w = *(const float4v*)(x + (size_t)t * HD + i);
    u.s[0] = f2bf(w.x); u.s[1] = f2bf(w.y); u.s[2] = f2bf(w.z); u.s[3] = f2bf(w.w);
  } else {
    u.s[0] = u.s[1] = u.s[2] = u.s[3] = 0;
  }
  *(uint2v*)(Xg + (size_t)p * HD + i) = u.v;
}

// ---------------- K4: GEMM1  Hmid = gelu(Xg @ w1[e] + b1[e]),  M=32 N=128 K=32, double-buffered ----------------
__global__ __launch_bounds__(256) void moe_gemm1(
    const unsigned short* __restrict__ Xg, const float* __restrict__ w1,
    const float* __restrict__ b1, unsigned short* __restrict__ Hmid,
    const int* __restrict__ tile_e, const int* __restrict__ tile_r,
    const int* __restrict__ ntiles) {
  int mt = blockIdx.y;
  if (mt >= ntiles[0]) return;                  // uniform across block
  int e = tile_e[mt], row0 = tile_r[mt];
  int n0 = blockIdx.x * 128;
  int tid = threadIdx.x;
  int wave = tid >> 5, lane = tid & 31;
  int half = lane >> 4, lr = lane & 15;
  __shared__ unsigned short As[2][32][40];      // double-buffered 32(M) x 32(K) bf16
  __shared__ unsigned short BsT[2][128][40];    // double-buffered transposed [N][K] bf16
  const float* wbase = w1 + (size_t)e * HD * FD + n0;
  const unsigned short* abase = Xg + (size_t)row0 * HD;
  v8f acc0 = {}, acc1 = {};
  int ar = tid >> 3, ac = (tid & 7) * 4;        // each thread async-copies 8B of the A tile
  unsigned a_loff[2] = { (unsigned)(size_t)(lds_ushort_t*)&As[0][ar][ac],
                         (unsigned)(size_t)(lds_ushort_t*)&As[1][ar][ac] };
  // fill: async bf16 A tile + fp32->bf16 transposed B tile (K-pairs packed as b32 stores)
  auto fill = [&](int buf, int kk) {
    asm volatile("global_load_async_to_lds_b64 %0, %1, off"
                 :: "v"(a_loff[buf]), "v"(abase + (size_t)ar * HD + kk + ac) : "memory");
#pragma unroll
    for (int i = 0; i < 2; i++) {
      int q = tid + i * 256;                    // 512 row-pair x col-quad work items
      int rp = (q >> 5) * 2, c4 = (q & 31) * 4;
      float4v w0 = *(const float4v*)(wbase + (size_t)(kk + rp) * FD + c4);
      float4v w1v = *(const float4v*)(wbase + (size_t)(kk + rp + 1) * FD + c4);
      PK2 p0, p1, p2, p3;
      p0.s[0] = f2bf(w0.x); p0.s[1] = f2bf(w1v.x);
      p1.s[0] = f2bf(w0.y); p1.s[1] = f2bf(w1v.y);
      p2.s[0] = f2bf(w0.z); p2.s[1] = f2bf(w1v.z);
      p3.s[0] = f2bf(w0.w); p3.s[1] = f2bf(w1v.w);
      *(unsigned int*)&BsT[buf][c4 + 0][rp] = p0.u;
      *(unsigned int*)&BsT[buf][c4 + 1][rp] = p1.u;
      *(unsigned int*)&BsT[buf][c4 + 2][rp] = p2.u;
      *(unsigned int*)&BsT[buf][c4 + 3][rp] = p3.u;
    }
    if (kk + 32 < HD)
      __builtin_prefetch(wbase + (size_t)(kk + 32) * FD + (tid & 31) * 4, 0, 1);
  };
  fill(0, 0);                                   // prologue
  for (int it = 0; it < HD / 32; ++it) {
    asm volatile("s_wait_asynccnt 0x0" ::: "memory");
    __syncthreads();                            // buffer `cur` ready for all waves
    int cur = it & 1;
    int kn = (it + 1) * 32;
    if (kn < HD) fill(cur ^ 1, kn);             // overlap next tile's loads with WMMAs
    Frag a0, a1, b;
    a0.q[0] = *(const uint4v*)&As[cur][lr][half * 8];          // K 0..7 / 8..15
    a0.q[1] = *(const uint4v*)&As[cur][lr][16 + half * 8];     // K 16..23 / 24..31
    a1.q[0] = *(const uint4v*)&As[cur][16 + lr][half * 8];
    a1.q[1] = *(const uint4v*)&As[cur][16 + lr][16 + half * 8];
    b.q[0]  = *(const uint4v*)&BsT[cur][wave * 16 + lr][half * 8];      // K 0..7 / 8..15
    b.q[1]  = *(const uint4v*)&BsT[cur][wave * 16 + lr][16 + half * 8];
    acc0 = __builtin_amdgcn_wmma_f32_16x16x32_bf16(false, a0.v, false, b.v,
                                                   (short)0, acc0, false, false);
    acc1 = __builtin_amdgcn_wmma_f32_16x16x32_bf16(false, a1.v, false, b.v,
                                                   (short)0, acc1, false, false);
  }
  int col = n0 + wave * 16 + lr;
  float bias = b1[(size_t)e * FD + col];
  CFrag c0, c1; c0.v = acc0; c1.v = acc1;
#pragma unroll
  for (int j = 0; j < 8; j++) {
    int m = half * 8 + j;                       // C layout: VGPR j -> M=j (+8 hi half-wave)
    Hmid[(size_t)(row0 + m) * FD + col]      = f2bf(geluf(c0.e[j] + bias));
    Hmid[(size_t)(row0 + 16 + m) * FD + col] = f2bf(geluf(c1.e[j] + bias));
  }
}

// ---------------- K5: GEMM2  out[token] = (Hmid @ w2[e] + b2[e]) * topw ----------------
__global__ __launch_bounds__(256) void moe_gemm2(
    const unsigned short* __restrict__ Hmid, const float* __restrict__ w2,
    const float* __restrict__ b2, const int* __restrict__ perm,
    const float* __restrict__ topw, float* __restrict__ out,
    const int* __restrict__ tile_e, const int* __restrict__ tile_r,
    const int* __restrict__ ntiles) {
  int mt = blockIdx.y;
  if (mt >= ntiles[0]) return;
  int e = tile_e[mt], row0 = tile_r[mt];
  int n0 = blockIdx.x * 128;
  int tid = threadIdx.x;
  int wave = tid >> 5, lane = tid & 31;
  int half = lane >> 4, lr = lane & 15;
  __shared__ unsigned short As[2][32][40];
  __shared__ unsigned short BsT[2][128][40];
  const float* wbase = w2 + (size_t)e * FD * HD + n0;
  const unsigned short* abase = Hmid + (size_t)row0 * FD;
  v8f acc0 = {}, acc1 = {};
  int ar = tid >> 3, ac = (tid & 7) * 4;
  unsigned a_loff[2] = { (unsigned)(size_t)(lds_ushort_t*)&As[0][ar][ac],
                         (unsigned)(size_t)(lds_ushort_t*)&As[1][ar][ac] };
  auto fill = [&](int buf, int kk) {
    asm volatile("global_load_async_to_lds_b64 %0, %1, off"
                 :: "v"(a_loff[buf]), "v"(abase + (size_t)ar * FD + kk + ac) : "memory");
#pragma unroll
    for (int i = 0; i < 2; i++) {
      int q = tid + i * 256;
      int rp = (q >> 5) * 2, c4 = (q & 31) * 4;
      float4v w0 = *(const float4v*)(wbase + (size_t)(kk + rp) * HD + c4);
      float4v w1v = *(const float4v*)(wbase + (size_t)(kk + rp + 1) * HD + c4);
      PK2 p0, p1, p2, p3;
      p0.s[0] = f2bf(w0.x); p0.s[1] = f2bf(w1v.x);
      p1.s[0] = f2bf(w0.y); p1.s[1] = f2bf(w1v.y);
      p2.s[0] = f2bf(w0.z); p2.s[1] = f2bf(w1v.z);
      p3.s[0] = f2bf(w0.w); p3.s[1] = f2bf(w1v.w);
      *(unsigned int*)&BsT[buf][c4 + 0][rp] = p0.u;
      *(unsigned int*)&BsT[buf][c4 + 1][rp] = p1.u;
      *(unsigned int*)&BsT[buf][c4 + 2][rp] = p2.u;
      *(unsigned int*)&BsT[buf][c4 + 3][rp] = p3.u;
    }
    if (kk + 32 < FD)
      __builtin_prefetch(wbase + (size_t)(kk + 32) * HD + (tid & 31) * 4, 0, 1);
  };
  fill(0, 0);
  for (int it = 0; it < FD / 32; ++it) {
    asm volatile("s_wait_asynccnt 0x0" ::: "memory");
    __syncthreads();
    int cur = it & 1;
    int kn = (it + 1) * 32;
    if (kn < FD) fill(cur ^ 1, kn);
    Frag a0, a1, b;
    a0.q[0] = *(const uint4v*)&As[cur][lr][half * 8];
    a0.q[1] = *(const uint4v*)&As[cur][lr][16 + half * 8];
    a1.q[0] = *(const uint4v*)&As[cur][16 + lr][half * 8];
    a1.q[1] = *(const uint4v*)&As[cur][16 + lr][16 + half * 8];
    b.q[0]  = *(const uint4v*)&BsT[cur][wave * 16 + lr][half * 8];
    b.q[1]  = *(const uint4v*)&BsT[cur][wave * 16 + lr][16 + half * 8];
    acc0 = __builtin_amdgcn_wmma_f32_16x16x32_bf16(false, a0.v, false, b.v,
                                                   (short)0, acc0, false, false);
    acc1 = __builtin_amdgcn_wmma_f32_16x16x32_bf16(false, a1.v, false, b.v,
                                                   (short)0, acc1, false, false);
  }
  int col = n0 + wave * 16 + lr;
  float bias = b2[(size_t)e * HD + col];
  CFrag c0, c1; c0.v = acc0; c1.v = acc1;
#pragma unroll
  for (int j = 0; j < 8; j++) {
    int m = half * 8 + j;
    int r0 = row0 + m, r1 = row0 + 16 + m;
    int t0 = perm[r0], t1 = perm[r1];
    if (t0 >= 0) out[(size_t)t0 * HD + col] = (c0.e[j] + bias) * topw[t0];
    if (t1 >= 0) out[(size_t)t1 * HD + col] = (c1.e[j] + bias) * topw[t1];
  }
}

// ---------------- K6: deterministic aux loss ----------------
__global__ void moe_aux(const int* __restrict__ counts, const float* __restrict__ probs,
                        float* __restrict__ out_aux) {
  int e = threadIdx.x;                          // 32 threads, lanes >= NE contribute 0
  float part = 0.f;
  if (e < NE) {
    float s = 0.f;
    for (int t = 0; t < TOK; t++) s += probs[(size_t)t * NE + e];
    part = (s / (float)TOK) * ((float)counts[e] / (float)TOK);
  }
  for (int off = 16; off >= 1; off >>= 1) part += __shfl_xor(part, off, 32);
  if (e == 0) out_aux[0] = LBWF * (float)NE * part;
}

extern "C" void kernel_launch(void* const* d_in, const int* in_sizes, int n_in,
                              void* d_out, int out_size, void* d_ws, size_t ws_size,
                              hipStream_t stream) {
  (void)in_sizes; (void)n_in; (void)out_size; (void)ws_size;
  const float* x  = (const float*)d_in[0];
  const float* rw = (const float*)d_in[1];
  const float* w1 = (const float*)d_in[2];
  const float* b1 = (const float*)d_in[3];
  const float* w2 = (const float*)d_in[4];
  const float* b2 = (const float*)d_in[5];
  float* out = (float*)d_out;

  char* ws = (char*)d_ws;
  size_t off = 0;
  auto walloc = [&](size_t bytes) -> void* {
    void* p = ws + off;
    off = (off + bytes + 255) & ~(size_t)255;
    return p;
  };
  float* topw    = (float*)walloc((size_t)TOK * 4);
  int*   topidx  = (int*)  walloc((size_t)TOK * 4);
  int*   counts  = (int*)  walloc(NE * 4);
  int*   cursor  = (int*)  walloc(NE * 4);
  int*   offsets = (int*)  walloc((NE + 1) * 4);
  int*   tile_e  = (int*)  walloc(MAXTILES * 4);
  int*   tile_r  = (int*)  walloc(MAXTILES * 4);
  int*   ntiles  = (int*)  walloc(16);
  int*   perm    = (int*)  walloc((size_t)PCAP * 4);
  float* probs   = (float*)walloc((size_t)TOK * NE * 4);
  unsigned short* Xg = (unsigned short*)walloc((size_t)PCAP * HD * 2);
  unsigned short* Hm = (unsigned short*)walloc((size_t)PCAP * FD * 2);

  moe_init  <<<4, 256, 0, stream>>>(counts, cursor, perm);
  moe_router<<<TOK / 4, 128, 0, stream>>>(x, rw, topw, topidx, counts, probs);
  moe_scan  <<<1, 256, 0, stream>>>(counts, offsets, tile_e, tile_r, ntiles,
                                    topidx, cursor, perm);
  moe_gather<<<PCAP, 256, 0, stream>>>(x, perm, Xg);
  moe_gemm1 <<<dim3(FD / 128, MAXTILES), 256, 0, stream>>>(Xg, w1, b1, Hm,
                                                           tile_e, tile_r, ntiles);
  moe_gemm2 <<<dim3(HD / 128, MAXTILES), 256, 0, stream>>>(Hm, w2, b2, perm, topw, out,
                                                           tile_e, tile_r, ntiles);
  moe_aux   <<<1, 32, 0, stream>>>(counts, probs, out + (size_t)TOK * HD);
}
